// AttentionBlock_42838003810499
// MI455X (gfx1250) — matrix operand order repaired and checked by reading, not compiled, not run on hardware
//
#include <hip/hip_runtime.h>
#include <hip/hip_bf16.h>

typedef _Float16 v16h __attribute__((ext_vector_type(16)));
typedef _Float16 v8h  __attribute__((ext_vector_type(8)));
typedef float    v8f  __attribute__((ext_vector_type(8)));

#define CDIM 768
#define FDIM 2304
#define NTOK 32768
#define NHEADS 12
#define HDIM 64

// -------- async global->LDS (gfx1250 ASYNCcnt path), with safe fallback ------
#if __has_builtin(__builtin_amdgcn_global_load_async_to_lds_b128)
#define HAS_ASYNC_LDS 1
#else
#define HAS_ASYNC_LDS 0
#endif

__device__ __forceinline__ void async_copy_b128(const _Float16* g, _Float16* l) {
#if HAS_ASYNC_LDS
    typedef int v4i_ __attribute__((vector_size(4 * sizeof(int))));
    typedef __attribute__((address_space(1))) v4i_* gp_t;
    typedef __attribute__((address_space(3))) v4i_* lp_t;
    __attribute__((address_space(1))) _Float16* g1 =
        (__attribute__((address_space(1))) _Float16*)(_Float16*)g;   // strip const, cast to AS1
    __attribute__((address_space(3))) _Float16* l3 =
        (__attribute__((address_space(3))) _Float16*)l;              // cast to AS3
    __builtin_amdgcn_global_load_async_to_lds_b128((gp_t)g1, (lp_t)l3, 0, 0);
#else
    *(v8h*)l = *(const v8h*)g;
#endif
}
__device__ __forceinline__ void async_wait0() {
#if HAS_ASYNC_LDS
    asm volatile("s_wait_asynccnt 0x0" ::: "memory");
#endif
}
__device__ __forceinline__ void async_wait1() {
#if HAS_ASYNC_LDS
    asm volatile("s_wait_asynccnt 0x1" ::: "memory");
#endif
}

// ---------------------------------------------------------------- f32 -> f16
__global__ void cvt_f16_kernel(const float* __restrict__ src, _Float16* __restrict__ dst, int n) {
    int i = blockIdx.x * 256 + threadIdx.x;
    if (i < n) dst[i] = (_Float16)src[i];
}

// ---------------------------------------------------------------- LayerNorm (f32 in, f16 out)
__global__ void ln_f32_kernel(const float* __restrict__ x, const float* __restrict__ g,
                              const float* __restrict__ bta, _Float16* __restrict__ y) {
    int row = blockIdx.x;
    const float* xr = x + (size_t)row * CDIM;
    int t = threadIdx.x;
    float v0 = xr[t], v1 = xr[t + 256], v2 = xr[t + 512];
    __shared__ float red[512];
    red[t] = v0 + v1 + v2;
    red[t + 256] = v0 * v0 + v1 * v1 + v2 * v2;
    __syncthreads();
    for (int off = 128; off > 0; off >>= 1) {
        if (t < off) { red[t] += red[t + off]; red[256 + t] += red[256 + t + off]; }
        __syncthreads();
    }
    float mean = red[0] * (1.0f / CDIM);
    float var  = red[256] * (1.0f / CDIM) - mean * mean;
    float rstd = rsqrtf(var + 1e-5f);
    _Float16* yr = y + (size_t)row * CDIM;
    yr[t]       = (_Float16)((v0 - mean) * rstd * g[t]       + bta[t]);
    yr[t + 256] = (_Float16)((v1 - mean) * rstd * g[t + 256] + bta[t + 256]);
    yr[t + 512] = (_Float16)((v2 - mean) * rstd * g[t + 512] + bta[t + 512]);
}

// ---------------------------------------------------------------- LayerNorm (f16 in, f16 out)
__global__ void ln_f16_kernel(const _Float16* __restrict__ x, const float* __restrict__ g,
                              const float* __restrict__ bta, _Float16* __restrict__ y) {
    int row = blockIdx.x;
    const _Float16* xr = x + (size_t)row * CDIM;
    int t = threadIdx.x;
    float v0 = (float)xr[t], v1 = (float)xr[t + 256], v2 = (float)xr[t + 512];
    __shared__ float red[512];
    red[t] = v0 + v1 + v2;
    red[t + 256] = v0 * v0 + v1 * v1 + v2 * v2;
    __syncthreads();
    for (int off = 128; off > 0; off >>= 1) {
        if (t < off) { red[t] += red[t + off]; red[256 + t] += red[256 + t + off]; }
        __syncthreads();
    }
    float mean = red[0] * (1.0f / CDIM);
    float var  = red[256] * (1.0f / CDIM) - mean * mean;
    float rstd = rsqrtf(var + 1e-5f);
    _Float16* yr = y + (size_t)row * CDIM;
    yr[t]       = (_Float16)((v0 - mean) * rstd * g[t]       + bta[t]);
    yr[t + 256] = (_Float16)((v1 - mean) * rstd * g[t + 256] + bta[t + 256]);
    yr[t + 512] = (_Float16)((v2 - mean) * rstd * g[t + 512] + bta[t + 512]);
}

// ---------------------------------------------------------------- blocked GEMM mainloop
// Block tile 128M x 64N (8 waves x 16 rows, shared B panel in LDS, double-buffered).
// B panel per K-step: 64 cols x 32 K f16, padded to 40 halves/row (bank-friendly, 16B aligned).
__device__ __forceinline__ void gemm_mainloop(const _Float16* __restrict__ A,
                                              const _Float16* __restrict__ Bw,
                                              _Float16 (*lds_b)[64][40],
                                              int m0, int n0, int lane, v8f* accs) {
    const int K = CDIM;
    int tid = threadIdx.x;
    int colL = tid >> 2;            // 0..63  cooperative B loader: col
    int seg  = (tid & 3) * 8;       // 0,8,16,24: 8-half segment within K-step
    const _Float16* gB = Bw + (size_t)(n0 + colL) * K + seg;
    _Float16* lB0 = &lds_b[0][colL][seg];
    _Float16* lB1 = &lds_b[1][colL][seg];

    int lrow = lane & 15;
    const _Float16* arow = A + (size_t)(m0 + lrow) * K + ((lane < 16) ? 0 : 16);
    int koffB = (lane < 16) ? 0 : 16;

    async_copy_b128(gB, lB0);                    // preload k0 = 0
    for (int k0 = 0; k0 < K; k0 += 32) {
        int buf = (k0 >> 5) & 1;
        bool more = (k0 + 32) < K;
        if (more) {
            async_copy_b128(gB + k0 + 32, buf ? lB0 : lB1);
            async_wait1();                        // current panel done (in-order)
        } else {
            async_wait0();
        }
        __syncthreads();                          // panel visible to all waves
        __builtin_prefetch(arow + k0 + 128, 0, 0);
        union { v16h v; v8h h[2]; } ua;
        ua.h[0] = *(const v8h*)(arow + k0);
        ua.h[1] = *(const v8h*)(arow + k0 + 8);
        v16h bb0 = *(const v16h*)(&lds_b[buf][ 0 + lrow][koffB]);
        v16h bb1 = *(const v16h*)(&lds_b[buf][16 + lrow][koffB]);
        v16h bb2 = *(const v16h*)(&lds_b[buf][32 + lrow][koffB]);
        v16h bb3 = *(const v16h*)(&lds_b[buf][48 + lrow][koffB]);
        accs[0] = __builtin_amdgcn_wmma_f32_16x16x32_f16(false, ua.v, false, bb0, (short)0, accs[0], false, false);
        accs[1] = __builtin_amdgcn_wmma_f32_16x16x32_f16(false, ua.v, false, bb1, (short)0, accs[1], false, false);
        accs[2] = __builtin_amdgcn_wmma_f32_16x16x32_f16(false, ua.v, false, bb2, (short)0, accs[2], false, false);
        accs[3] = __builtin_amdgcn_wmma_f32_16x16x32_f16(false, ua.v, false, bb3, (short)0, accs[3], false, false);
        __syncthreads();                          // all reads done before next overwrite
    }
}

// ---------------------------------------------------------------- QKV GEMM: [NTOK,768] x [2304,768]^T
__global__ void gemm_qkv_kernel(const _Float16* __restrict__ A, const _Float16* __restrict__ Bw,
                                const float* __restrict__ bias, _Float16* __restrict__ out) {
    const int N = FDIM, nN = N / 64;
    __shared__ __align__(16) _Float16 lds_b[2][64][40];
    int lane = threadIdx.x & 31, wave = threadIdx.x >> 5;
    int mb = blockIdx.x / nN, nb = blockIdx.x % nN;
    int m0 = mb * 128 + wave * 16, n0 = nb * 64;
    v8f accs[4] = {};
    gemm_mainloop(A, Bw, lds_b, m0, n0, lane, accs);
    int rowoff = (lane < 16) ? 0 : 8;
    for (int j = 0; j < 4; j++) {
        int col = n0 + j * 16 + (lane & 15);
        float bia = bias[col];
        for (int r = 0; r < 8; r++)
            out[(size_t)(m0 + rowoff + r) * N + col] = (_Float16)(accs[j][r] + bia);
    }
}

// ---------------------------------------------------------------- Out-proj GEMM + bias + residual
__global__ void gemm_out_kernel(const _Float16* __restrict__ A, const _Float16* __restrict__ Bw,
                                const float* __restrict__ bias, const float* __restrict__ resid,
                                float* __restrict__ out) {
    const int N = CDIM, nN = N / 64;
    __shared__ __align__(16) _Float16 lds_b[2][64][40];
    int lane = threadIdx.x & 31, wave = threadIdx.x >> 5;
    int mb = blockIdx.x / nN, nb = blockIdx.x % nN;
    int m0 = mb * 128 + wave * 16, n0 = nb * 64;
    v8f accs[4] = {};
    gemm_mainloop(A, Bw, lds_b, m0, n0, lane, accs);
    int rowoff = (lane < 16) ? 0 : 8;
    for (int j = 0; j < 4; j++) {
        int col = n0 + j * 16 + (lane & 15);
        float bia = bias[col];
        for (int r = 0; r < 8; r++) {
            size_t idx = (size_t)(m0 + rowoff + r) * N + col;
            out[idx] = accs[j][r] + bia + resid[idx];
        }
    }
}

// ---------------------------------------------------------------- Attention over time axis
// One wave per (b,h,w,head) instance: scores(16x16)=2 WMMAs, softmax via lane
// shuffles, P@V = 4 WMMAs with K padded 16->32 with zeros. P and V transposed via LDS.
__global__ void attn_kernel(const _Float16* __restrict__ qkv, _Float16* __restrict__ o) {
    __shared__ _Float16 lds_p[8][16][16];
    __shared__ __align__(16) _Float16 lds_v[8][16][64];
    int lane = threadIdx.x & 31, wave = threadIdx.x >> 5;
    int inst = blockIdx.x * 8 + wave;
    int nh = inst % NHEADS; int tmpi = inst / NHEADS;
    int wi = tmpi % 32; tmpi /= 32;
    int hi = tmpi % 32; int bi = tmpi / 32;
    size_t base_tok = (size_t)bi * 16 * 1024 + (size_t)hi * 32 + wi;   // + t*1024
    int lrow = lane & 15;
    size_t tokL = base_tok + (size_t)lrow * 1024;
    const _Float16* qrow = qkv + tokL * FDIM + nh * 192 + ((lane < 16) ? 0 : 16);
    const _Float16* krow = qkv + tokL * FDIM + nh * 192 + 64 + ((lane < 16) ? 0 : 16);
    const _Float16* vrow = qkv + tokL * FDIM + nh * 192 + 128;

    // scores S = q k^T, K-dim = head_dim 64 -> two 16x16x32 WMMAs
    v8f sacc = {};
    for (int d0 = 0; d0 < HDIM; d0 += 32) {
        union { v16h v; v8h h[2]; } ua;
        ua.h[0] = *(const v8h*)(qrow + d0);
        ua.h[1] = *(const v8h*)(qrow + d0 + 8);
        v16h kb = *(const v16h*)(krow + d0);
        sacc = __builtin_amdgcn_wmma_f32_16x16x32_f16(false, ua.v, false, kb, (short)0, sacc, false, false);
    }
    // softmax over the s axis (16 lanes per half-wave hold one row each)
    float p[8];
    for (int r = 0; r < 8; r++) {
        float xv = sacc[r] * 0.125f;       // 1/sqrt(64)
        float m = xv;
        for (int msk = 1; msk < 16; msk <<= 1) m = fmaxf(m, __shfl_xor(m, msk, 32));
        float e = __expf(xv - m);
        float s = e;
        for (int msk = 1; msk < 16; msk <<= 1) s += __shfl_xor(s, msk, 32);
        p[r] = e / s;
    }
    // stage V (16x64) and P (16x16, as f16) in LDS for the layout transpose
    for (int i = 0; i < 2; i++) {
        int ch = (lane >> 4) + 2 * i;   // 0..3 chunks of 16 halves
        *(v16h*)(&lds_v[wave][lrow][ch * 16]) = *(const v16h*)(vrow + ch * 16);
    }
    int rowoff = (lane < 16) ? 0 : 8;
    for (int r = 0; r < 8; r++) lds_p[wave][rowoff + r][lrow] = (_Float16)p[r];
    __syncthreads();

    // A-frag of P: lane<16 -> K(s)=0..15, lane>=16 (K=16..31) zero-padded
    v16h pa = {};
    int sb = (lane < 16) ? 0 : 8;
    if (lane < 16) for (int j = 0; j < 16; j++) pa[j] = lds_p[wave][lrow][j];

    for (int jd = 0; jd < 4; jd++) {
        // B-frag of V: lanes 0-15 hold K=0..15 for column d; lanes 16-31 (K=16..31) zero
        v16h pb = {};
        if (lane < 16) {
            int d = jd * 16 + lane;
            for (int s = 0; s < 16; s++) pb[s] = lds_v[wave][s][d];
        }
        v8f oacc = {};
        oacc = __builtin_amdgcn_wmma_f32_16x16x32_f16(false, pa, false, pb, (short)0, oacc, false, false);
        int col = nh * HDIM + jd * 16 + lrow;
        for (int r = 0; r < 8; r++) {
            size_t tok = base_tok + (size_t)(rowoff + r) * 1024;
            o[tok * CDIM + col] = (_Float16)oacc[r];
        }
    }
}

// ----------------------------------------------------------------
extern "C" void kernel_launch(void* const* d_in, const int* in_sizes, int n_in,
                              void* d_out, int out_size, void* d_ws, size_t ws_size,
                              hipStream_t stream) {
    const float* x    = (const float*)d_in[0];
    const float* ln1w = (const float*)d_in[1];
    const float* ln1b = (const float*)d_in[2];
    const float* Wqkv = (const float*)d_in[3];
    const float* bqkv = (const float*)d_in[4];
    const float* ln2w = (const float*)d_in[5];
    const float* ln2b = (const float*)d_in[6];
    const float* Wout = (const float*)d_in[7];
    const float* bout = (const float*)d_in[8];
    float* out = (float*)d_out;

    char* ws = (char*)d_ws;
    const size_t szW1  = (size_t)FDIM * CDIM * 2;   // 3.375 MB
    const size_t szW2  = (size_t)CDIM * CDIM * 2;   // 1.125 MB
    const size_t szY   = (size_t)NTOK * CDIM * 2;   // 48 MB
    const size_t szQKV = (size_t)NTOK * FDIM * 2;   // 144 MB
    _Float16* W1h  = (_Float16*)(ws);
    _Float16* W2h  = (_Float16*)(ws + szW1);
    _Float16* yh   = (_Float16*)(ws + szW1 + szW2);            // reused for LN2 output
    _Float16* qkvh = (_Float16*)(ws + szW1 + szW2 + szY);
    _Float16* oh   = (_Float16*)(ws + szW1 + szW2 + szY + szQKV);

    int nW1 = FDIM * CDIM, nW2 = CDIM * CDIM;
    cvt_f16_kernel<<<(nW1 + 255) / 256, 256, 0, stream>>>(Wqkv, W1h, nW1);
    cvt_f16_kernel<<<(nW2 + 255) / 256, 256, 0, stream>>>(Wout, W2h, nW2);

    ln_f32_kernel<<<NTOK, 256, 0, stream>>>(x, ln1w, ln1b, yh);

    // (NTOK/128) * (FDIM/64) block tiles of 128x64
    gemm_qkv_kernel<<<(NTOK / 128) * (FDIM / 64), 256, 0, stream>>>(yh, W1h, bqkv, qkvh);

    // 2*32*32*12 = 24576 head-instances, 8 per block
    attn_kernel<<<24576 / 8, 256, 0, stream>>>(qkvh, oh);

    ln_f16_kernel<<<NTOK, 256, 0, stream>>>(oh, ln2w, ln2b, yh);

    gemm_out_kernel<<<(NTOK / 128) * (CDIM / 64), 256, 0, stream>>>(yh, W2h, bout, x, out);
}
